// SwinTransformerBlock_67963562492511
// MI455X (gfx1250) — matrix operand order repaired and checked by reading, not compiled, not run on hardware
//
#include <hip/hip_runtime.h>
#include <hip/hip_bf16.h>

// ---------------------------------------------------------------------------
// Swin Transformer block for MI455X (gfx1250): wave32, WMMA f16 16x16x32.
// All matmul B-operands are stored transposed (N-major) so every fragment
// load is two contiguous 16-byte runs per lane -> global_load_b128 / ds_load_b128.
// Kernel 1: one workgroup (256 thr = 8 waves) per 8x8 window -> LN1, QKV,
//           attention (bias+mask+softmax), P@V, proj, residual. LDS = 176 KB.
// Kernel 2: one workgroup per 64-token tile -> LN2, MLP(512)+GELU, residual.
// ---------------------------------------------------------------------------

typedef __attribute__((ext_vector_type(16))) _Float16 v16h;
typedef __attribute__((ext_vector_type(8)))  float    v8f;

#define WMMA_F16(a, b, c) \
  __builtin_amdgcn_wmma_f32_16x16x32_f16(false, (a), false, (b), (short)0, (c), false, false)

// A-matrix 16x32 f16 per-lane K interleave (ISA 7.12.2):
// lanes 0-15: K 0..7 then 16..23 ; lanes 16-31: K 8..15 then 24..31
__device__ __forceinline__ int ka_map(int i, int lh) {
  return ((i >> 3) << 4) + (lh << 3) + (i & 7);
}

// A fragment from row-major X (M x K), ld in halves. Two contiguous 8-half
// runs per lane -> 2x b128 loads.
__device__ __forceinline__ v16h frag_a(const _Float16* X, int ld, int rbase, int kbase, int lane) {
  v16h a;
  int m = rbase + (lane & 15);
  int lh = lane >> 4;
#pragma unroll
  for (int i = 0; i < 16; ++i) a[i] = X[m * ld + kbase + ka_map(i, lh)];
  return a;
}

// B fragment where the matrix is available transposed: B[k][n] = X[n][k],
// X row-major (N x K). Two contiguous 8-half runs per lane -> 2x b128 loads.
// lanes 0-15: K 0..7 then 16..23 ; lanes 16-31: K 8..15 then 24..31
__device__ __forceinline__ v16h frag_bt(const _Float16* X, int ld, int kbase, int nbase, int lane) {
  v16h b;
  const _Float16* p = X + (nbase + (lane & 15)) * ld + kbase + (lane >> 4) * 8;
#pragma unroll
  for (int i = 0; i < 16; ++i) b[i] = p[((i >> 3) << 4) + (i & 7)];
  return b;
}

__device__ __forceinline__ int region_label(int r, int c) {
  int lr = (r < 248) ? 0 : ((r < 252) ? 1 : 2);
  int lc = (c < 248) ? 0 : ((c < 252) ? 1 : 2);
  return lr * 3 + lc;
}

// ---------------------------------------------------------------------------
// Kernel 0: convert fp32 weights to f16 TRANSPOSED (N-major) into workspace.
// layout (halves):
//   qkvwt [0,      49152) : (384 x 128)  qkvwt[n*128+k] = qkv_w[k*384+n]
//   projwt[49152,  65536) : (128 x 128)  projwt[n*128+k] = proj_w[k*128+n]
//   w1t   [65536, 131072) : (512 x 128)  w1t[n*128+k]   = w1[k*512+n]
//   w2t   [131072,196608) : (128 x 512)  w2t[n*512+k]   = w2[k*128+n]
// ---------------------------------------------------------------------------
__global__ void convert_weights_kernel(const float* __restrict__ qkvw,
                                       const float* __restrict__ projw,
                                       const float* __restrict__ w1,
                                       const float* __restrict__ w2,
                                       _Float16* __restrict__ ws) {
  int i = blockIdx.x * 256 + threadIdx.x;
  float v;
  if (i < 49152) {
    int n = i >> 7, k = i & 127;
    v = qkvw[k * 384 + n];
  } else if (i < 65536) {
    int j = i - 49152, n = j >> 7, k = j & 127;
    v = projw[k * 128 + n];
  } else if (i < 131072) {
    int j = i - 65536, n = j >> 7, k = j & 127;
    v = w1[k * 512 + n];
  } else {
    int j = i - 131072, n = j >> 9, k = j & 511;
    v = w2[k * 128 + n];
  }
  ws[i] = (_Float16)v;
}

// ---------------------------------------------------------------------------
// Kernel 1: fused window attention.  blockIdx.x = window id (4096).
// Dynamic LDS (180224 B): xn 64x128 f16 (reused as O) | qkv 64x384 f16 |
//                         S 4x64x64 f32 | P 4x64x64 f16 | vt 4x32x64 f16
// ---------------------------------------------------------------------------
__global__ __launch_bounds__(256) void swin_attn_kernel(
    const float* __restrict__ x, const float* __restrict__ g1, const float* __restrict__ be1,
    const _Float16* __restrict__ qkvwt, const float* __restrict__ qkvb,
    const _Float16* __restrict__ projwt, const float* __restrict__ projb,
    const float* __restrict__ rpb, float* __restrict__ out) {
  extern __shared__ char smem[];
  _Float16* xn  = (_Float16*)(smem);            // 16384 B
  _Float16* qkv = (_Float16*)(smem + 16384);    // 49152 B (q,k row-major; v unused)
  float*    S   = (float*)   (smem + 65536);    // 65536 B
  _Float16* P   = (_Float16*)(smem + 131072);   // 32768 B
  _Float16* vt  = (_Float16*)(smem + 163840);   // 16384 B: [head][d(32)][m(64)]

  const int tid  = threadIdx.x;
  const int lane = tid & 31;
  const int wave = tid >> 5;
  const int wid  = blockIdx.x;
  const int bb   = wid >> 10;
  const int wrem = wid & 1023;
  const int wh   = wrem >> 5, ww = wrem & 31;

  // warm L2 for the f16 transposed weights (global_prefetch_b8)
  __builtin_prefetch(qkvwt + tid * 128, 0, 1);
  __builtin_prefetch(projwt + tid * 64, 0, 1);

  // ---- stage 1: load shifted window + LayerNorm -> xn (f16). 4 threads/token.
  {
    int t = tid >> 2, part = tid & 3;
    int th = t >> 3, tw = t & 7;
    int ph = (wh * 8 + th + 4) & 255;                 // cyclic shift by SS=4
    int pw = (ww * 8 + tw + 4) & 255;
    const float* xp = x + ((size_t)((bb * 256 + ph) * 256 + pw)) * 128 + part * 32;
    float v[32];
    float s = 0.f, ss = 0.f;
#pragma unroll
    for (int i = 0; i < 8; ++i) {
      float4 f = ((const float4*)xp)[i];
      v[i * 4 + 0] = f.x; v[i * 4 + 1] = f.y; v[i * 4 + 2] = f.z; v[i * 4 + 3] = f.w;
      s  += f.x + f.y + f.z + f.w;
      ss += f.x * f.x + f.y * f.y + f.z * f.z + f.w * f.w;
    }
    s  += __shfl_xor(s, 1, 4);  s  += __shfl_xor(s, 2, 4);
    ss += __shfl_xor(ss, 1, 4); ss += __shfl_xor(ss, 2, 4);
    float mean = s * (1.f / 128.f);
    float var  = ss * (1.f / 128.f) - mean * mean;
    float rstd = rsqrtf(var + 1.001e-05f);
#pragma unroll
    for (int i = 0; i < 32; ++i) {
      int c = part * 32 + i;
      xn[t * 128 + c] = (_Float16)((v[i] - mean) * rstd * g1[c] + be1[c]);
    }
  }
  __syncthreads();

  // ---- stage 2: QKV = xn(64x128) @ W(128x384) + b.
  // q (ct 0..7) scaled by HD^-0.5; k (ct 8..15) row-major into qkv;
  // v (ct 16..23) written TRANSPOSED into vt[head][d][m].
  for (int j = 0; j < 12; ++j) {
    int tt = wave * 12 + j;
    int rt = tt / 24, ct = tt % 24;
    v8f acc = {};
#pragma unroll
    for (int ks = 0; ks < 4; ++ks) {
      v16h a = frag_a(xn, 128, rt * 16, ks * 32, lane);
      v16h b = frag_bt(qkvwt, 128, ks * 32, ct * 16, lane);
      acc = WMMA_F16(a, b, acc);
    }
    int lh = lane >> 4, col = ct * 16 + (lane & 15);
    float bias = qkvb[col];
    if (ct < 8) {                                   // q: scale, row-major
#pragma unroll
      for (int r = 0; r < 8; ++r) {
        int row = rt * 16 + r + lh * 8;
        qkv[row * 384 + col] = (_Float16)((acc[r] + bias) * 0.17677669529663687f);
      }
    } else if (ct < 16) {                           // k: row-major
#pragma unroll
      for (int r = 0; r < 8; ++r) {
        int row = rt * 16 + r + lh * 8;
        qkv[row * 384 + col] = (_Float16)(acc[r] + bias);
      }
    } else {                                        // v: transposed vt[h][d][m]
      int cm = col - 256;
      int h = cm >> 5, d = cm & 31;
#pragma unroll
      for (int r = 0; r < 8; ++r) {
        int row = rt * 16 + r + lh * 8;             // m (token)
        vt[h * 2048 + d * 64 + row] = (_Float16)(acc[r] + bias);
      }
    }
  }
  __syncthreads();

  // ---- stage 3: S = q @ k^T + rel-pos bias + shift mask (per head).
  // head-dim 32 == WMMA K: one v_wmma per 16x16 tile. 2 waves per head.
  {
    int h = wave >> 1;
    for (int j = 0; j < 8; ++j) {
      int idx = (wave & 1) * 8 + j;
      int rt = idx >> 2, ct = idx & 3;
      v16h a = frag_a(qkv, 384, rt * 16, h * 32, lane);          // q rows
      v16h b = frag_bt(qkv, 384, 128 + h * 32, ct * 16, lane);   // k rows as B^T
      v8f acc = {};
      acc = WMMA_F16(a, b, acc);
      int lh = lane >> 4, m = ct * 16 + (lane & 15);
      int jh = m >> 3, jw = m & 7;
      int labm = region_label(wh * 8 + jh, ww * 8 + jw);
#pragma unroll
      for (int r = 0; r < 8; ++r) {
        int n  = rt * 16 + r + lh * 8;
        int ih = n >> 3, iw = n & 7;
        float bias = rpb[(((ih - jh + 7) * 15) + (iw - jw + 7)) * 4 + h];
        float mask = (region_label(wh * 8 + ih, ww * 8 + iw) == labm) ? 0.f : -100.f;
        S[(h * 64 + n) * 64 + m] = acc[r] + bias + mask;
      }
    }
  }
  __syncthreads();

  // ---- stage 4: softmax, one row per thread (4 heads x 64 rows = 256).
  {
    int h = tid >> 6, n = tid & 63;
    const float* row = S + (h * 64 + n) * 64;
    float mx = -1e30f;
#pragma unroll
    for (int m = 0; m < 64; ++m) mx = fmaxf(mx, row[m]);
    float sum = 0.f;
#pragma unroll
    for (int m = 0; m < 64; ++m) sum += __expf(row[m] - mx);
    float inv = 1.f / sum;
    _Float16* pr = P + (h * 64 + n) * 64;
#pragma unroll
    for (int m = 0; m < 64; ++m) pr[m] = (_Float16)(__expf(row[m] - mx) * inv);
  }
  __syncthreads();

  // ---- stage 5: O = P @ V via vt (contiguous B frags), into free xn region.
  _Float16* o = xn;
  for (int j = 0; j < 4; ++j) {
    int tt = wave * 4 + j;
    int h = tt >> 3, rem = tt & 7;
    int rt = rem >> 1, ct = rem & 1;
    v8f acc = {};
#pragma unroll
    for (int ks = 0; ks < 2; ++ks) {
      v16h a = frag_a(P + h * 4096, 64, rt * 16, ks * 32, lane);
      v16h b = frag_bt(vt + h * 2048, 64, ks * 32, ct * 16, lane);  // V^T rows=d
      acc = WMMA_F16(a, b, acc);
    }
    int lh = lane >> 4, col = h * 32 + ct * 16 + (lane & 15);
#pragma unroll
    for (int r = 0; r < 8; ++r) {
      int row = rt * 16 + r + lh * 8;
      o[row * 128 + col] = (_Float16)acc[r];
    }
  }
  __syncthreads();

  // ---- stage 6: proj + un-shift + residual -> out1 (fp32).
  for (int j = 0; j < 4; ++j) {
    int tt = wave * 4 + j;
    int rt = tt >> 3, ct = tt & 7;
    v8f acc = {};
#pragma unroll
    for (int ks = 0; ks < 4; ++ks) {
      v16h a = frag_a(o, 128, rt * 16, ks * 32, lane);
      v16h b = frag_bt(projwt, 128, ks * 32, ct * 16, lane);
      acc = WMMA_F16(a, b, acc);
    }
    int lh = lane >> 4, col = ct * 16 + (lane & 15);
    float pb = projb[col];
#pragma unroll
    for (int r = 0; r < 8; ++r) {
      int t = rt * 16 + r + lh * 8;
      int th = t >> 3, tw = t & 7;
      int ph = (wh * 8 + th + 4) & 255;
      int pw = (ww * 8 + tw + 4) & 255;
      size_t gi = ((size_t)((bb * 256 + ph) * 256 + pw)) * 128 + col;
      out[gi] = x[gi] + acc[r] + pb;
    }
  }
}

// ---------------------------------------------------------------------------
// Kernel 2: fused LN2 + MLP + residual. blockIdx.x = 64-token tile (4096).
// Dynamic LDS (81920 B): xn 64x128 f16 | h 64x512 f16
// ---------------------------------------------------------------------------
__global__ __launch_bounds__(256) void swin_mlp_kernel(
    const float* __restrict__ g2, const float* __restrict__ be2,
    const _Float16* __restrict__ w1t, const float* __restrict__ b1v,
    const _Float16* __restrict__ w2t, const float* __restrict__ b2v,
    float* __restrict__ out) {
  extern __shared__ char smem[];
  _Float16* xn = (_Float16*)smem;             // 16384 B
  _Float16* hh = (_Float16*)(smem + 16384);   // 65536 B

  const int tid = threadIdx.x, lane = tid & 31, wave = tid >> 5;
  const size_t tok0 = (size_t)blockIdx.x * 64;

  __builtin_prefetch(w1t + tid * 256, 0, 1);
  __builtin_prefetch(w2t + tid * 256, 0, 1);

  // ---- LN2 over the attention residual (reads d_out written by kernel 1)
  {
    int t = tid >> 2, part = tid & 3;
    const float* xp = out + (tok0 + t) * 128 + part * 32;
    float v[32];
    float s = 0.f, ss = 0.f;
#pragma unroll
    for (int i = 0; i < 8; ++i) {
      float4 f = ((const float4*)xp)[i];
      v[i * 4 + 0] = f.x; v[i * 4 + 1] = f.y; v[i * 4 + 2] = f.z; v[i * 4 + 3] = f.w;
      s  += f.x + f.y + f.z + f.w;
      ss += f.x * f.x + f.y * f.y + f.z * f.z + f.w * f.w;
    }
    s  += __shfl_xor(s, 1, 4);  s  += __shfl_xor(s, 2, 4);
    ss += __shfl_xor(ss, 1, 4); ss += __shfl_xor(ss, 2, 4);
    float mean = s * (1.f / 128.f);
    float var  = ss * (1.f / 128.f) - mean * mean;
    float rstd = rsqrtf(var + 1.001e-05f);
#pragma unroll
    for (int i = 0; i < 32; ++i) {
      int c = part * 32 + i;
      xn[t * 128 + c] = (_Float16)((v[i] - mean) * rstd * g2[c] + be2[c]);
    }
  }
  __syncthreads();

  // ---- GEMM1: (64x128) @ (128x512) + b1, exact GELU -> hh
  for (int j = 0; j < 16; ++j) {
    int tt = wave * 16 + j;
    int rt = tt >> 5, ct = tt & 31;
    v8f acc = {};
#pragma unroll
    for (int ks = 0; ks < 4; ++ks) {
      v16h a = frag_a(xn, 128, rt * 16, ks * 32, lane);
      v16h b = frag_bt(w1t, 128, ks * 32, ct * 16, lane);
      acc = WMMA_F16(a, b, acc);
    }
    int lh = lane >> 4, col = ct * 16 + (lane & 15);
    float bias = b1v[col];
#pragma unroll
    for (int r = 0; r < 8; ++r) {
      float u = acc[r] + bias;
      float g = 0.5f * u * (1.f + erff(u * 0.70710678118654752f));
      hh[(rt * 16 + r + lh * 8) * 512 + col] = (_Float16)g;
    }
  }
  __syncthreads();

  // ---- GEMM2: (64x512) @ (512x128) + b2 + residual -> out
  for (int j = 0; j < 4; ++j) {
    int tt = wave * 4 + j;
    int rt = tt >> 3, ct = tt & 7;
    v8f acc = {};
#pragma unroll
    for (int ks = 0; ks < 16; ++ks) {
      v16h a = frag_a(hh, 512, rt * 16, ks * 32, lane);
      v16h b = frag_bt(w2t, 512, ks * 32, ct * 16, lane);
      acc = WMMA_F16(a, b, acc);
    }
    int lh = lane >> 4, col = ct * 16 + (lane & 15);
    float bias = b2v[col];
#pragma unroll
    for (int r = 0; r < 8; ++r) {
      size_t gi = (tok0 + rt * 16 + r + lh * 8) * 128 + col;
      out[gi] = out[gi] + acc[r] + bias;   // same thread reads then writes gi
    }
  }
}

// ---------------------------------------------------------------------------
extern "C" void kernel_launch(void* const* d_in, const int* in_sizes, int n_in,
                              void* d_out, int out_size, void* d_ws, size_t ws_size,
                              hipStream_t stream) {
  (void)in_sizes; (void)n_in; (void)out_size; (void)ws_size;
  const float* x     = (const float*)d_in[0];
  const float* g1    = (const float*)d_in[1];
  const float* be1   = (const float*)d_in[2];
  const float* qkvw  = (const float*)d_in[3];
  const float* qkvb  = (const float*)d_in[4];
  const float* projw = (const float*)d_in[5];
  const float* projb = (const float*)d_in[6];
  const float* rpb   = (const float*)d_in[7];
  const float* g2    = (const float*)d_in[8];
  const float* be2   = (const float*)d_in[9];
  const float* w1    = (const float*)d_in[10];
  const float* b1v   = (const float*)d_in[11];
  const float* w2    = (const float*)d_in[12];
  const float* b2v   = (const float*)d_in[13];
  float* out = (float*)d_out;
  _Float16* wsh = (_Float16*)d_ws;

  // 196608 f16 weight elements total, stored transposed
  convert_weights_kernel<<<768, 256, 0, stream>>>(qkvw, projw, w1, w2, wsh);

  // one workgroup per window; 176 KB dynamic LDS (fits CDNA5's 320 KB WGP LDS)
  swin_attn_kernel<<<4096, 256, 180224, stream>>>(
      x, g1, be1, wsh, qkvb, wsh + 49152, projb, rpb, out);

  // one workgroup per 64-token tile; 80 KB dynamic LDS
  swin_mlp_kernel<<<4096, 256, 81920, stream>>>(
      g2, be2, wsh + 65536, b1v, wsh + 131072, b2v, out);
}
